// BinaryMatchingPursuit_37194416783917
// MI455X (gfx1250) — matrix operand order repaired and checked by reading, not compiled, not run on hardware
//
#include <hip/hip_runtime.h>
#include <hip/hip_bf16.h>

#define NB    4096
#define INF   512
#define OUTF  1024
#define KACT  11
#define KIN   6
#define KSTEPS (INF / 32)   // 16 K-steps of 32

typedef __attribute__((ext_vector_type(16))) _Float16 v16h;
typedef __attribute__((ext_vector_type(8)))  _Float16 v8h;
typedef __attribute__((ext_vector_type(8)))  float    v8f;
typedef __attribute__((ext_vector_type(4)))  unsigned int u32x4;
typedef __attribute__((ext_vector_type(8)))  int      i32x8;
typedef __attribute__((ext_vector_type(4)))  int      i32x4;

#if defined(__HIP_DEVICE_COMPILE__) && defined(__has_builtin)
#if __has_builtin(__builtin_amdgcn_tensor_load_to_lds) && __has_builtin(__builtin_amdgcn_s_wait_tensorcnt)
#define HAS_TDM 1
#endif
#endif
#ifndef HAS_TDM
#define HAS_TDM 0
#endif

// ---------------- wave32 helpers ----------------
__device__ __forceinline__ void wave_argmax(float &v, int &idx) {
  #pragma unroll
  for (int off = 16; off > 0; off >>= 1) {
    float v2 = __shfl_xor(v, off, 32);
    int   i2 = __shfl_xor(idx, off, 32);
    if (v2 > v || (v2 == v && i2 < idx)) { v = v2; idx = i2; }
  }
}

__device__ __forceinline__ int wave_sum(int v) {
  #pragma unroll
  for (int off = 16; off > 0; off >>= 1) v += __shfl_xor(v, off, 32);
  return v;
}

// pack (value, index) so u64-max == (max value, tie -> smallest index)
__device__ __forceinline__ unsigned long long pack_key(float v, int n) {
  unsigned u = __float_as_uint(v);
  u = (u & 0x80000000u) ? ~u : (u | 0x80000000u);   // order-preserving float->uint
  return ((unsigned long long)u << 32) | (unsigned long long)(0xFFFFFFFFu - (unsigned)n);
}

#if HAS_TDM
typedef const void __attribute__((address_space(3))) * lds_cptr;
__device__ __forceinline__ unsigned lds_offset(const void* p) {
  return (unsigned)(unsigned long long)(lds_cptr)p;   // AS3 ptr value == LDS byte offset
}

// D# per ISA ch.8: 2D tensor [OUTF x INF] f16, tile 32 rows x 32 halves -> LDS.
__device__ __forceinline__ void tdm_load_B(const _Float16* gsrc, unsigned ldsOff) {
  unsigned long long ga = (unsigned long long)(size_t)gsrc;
  u32x4 g0;
  g0[0] = 1u;                                   // count=1, user descriptor
  g0[1] = ldsOff;                               // lds_addr
  g0[2] = (unsigned)(ga & 0xFFFFFFFFu);         // global_addr[31:0]
  g0[3] = (unsigned)((ga >> 32) & 0x01FFFFFFu)  // global_addr[56:32]
        | (2u << 30);                           // type = 2 ("image")
  i32x8 g1;
  g1[0] = (int)(1u << 16);                      // workgroup_mask=0, data_size=1 (2B)
  g1[1] = (int)(((unsigned)INF & 0xFFFFu) << 16);        // tensor_dim0 lo -> [63:48]
  g1[2] = (int)((((unsigned)INF >> 16) & 0xFFFFu)
        | (((unsigned)OUTF & 0xFFFFu) << 16));           // dim0 hi, tensor_dim1 lo
  g1[3] = (int)((((unsigned)OUTF >> 16) & 0xFFFFu)
        | (32u << 16));                                  // dim1 hi, tile_dim0=32
  g1[4] = 32;                                   // tile_dim1=32, tile_dim2=0
  g1[5] = INF;                                  // tensor_dim0_stride lo
  g1[6] = 0;                                    // stride hi, dim1_stride lo
  g1[7] = 0;
  i32x4 z4 = {0, 0, 0, 0};
  i32x8 z8 = {0, 0, 0, 0, 0, 0, 0, 0};
  // 6-arg form (amdgpu-toolchain / clang-23 lane)
  __builtin_amdgcn_tensor_load_to_lds(g0, g1, z4, z4, z8, 0);
}
#endif

// ---------------- init kernels ----------------
__global__ void k_init_misc(int* maxNZ) {
  if (threadIdx.x == 0 && blockIdx.x == 0) *maxNZ = 0;
}

__global__ __launch_bounds__(256) void k_init_keys(unsigned long long* __restrict__ rowKey) {
  rowKey[blockIdx.x * 256 + threadIdx.x] = 0ull;
}

__global__ __launch_bounds__(256) void k_count_rows(const float* __restrict__ x,
                                                    int* __restrict__ maxNZ) {
  const int row  = blockIdx.x * 8 + (threadIdx.x >> 5);
  const int lane = threadIdx.x & 31;
  int cnt = 0;
  for (int j = lane; j < INF; j += 32)
    cnt += (x[(size_t)row * INF + j] != 0.0f) ? 1 : 0;
  cnt = wave_sum(cnt);
  if (lane == 0) atomicMax(maxNZ, cnt);
}

__global__ __launch_bounds__(256) void k_init_enc(float* __restrict__ encoded) {
  encoded[blockIdx.x * 256 + threadIdx.x] = 0.0f;
}

__global__ __launch_bounds__(256) void k_init_x(const float* __restrict__ x,
                                                float* __restrict__ xr,
                                                _Float16* __restrict__ Ah) {
  const int i = blockIdx.x * 256 + threadIdx.x;
  xr[i] = 0.0f;
  Ah[i] = (_Float16)(2.0f * x[i]);
}

__global__ __launch_bounds__(256) void k_conv_w(const float* __restrict__ W,
                                                _Float16* __restrict__ Wh) {
  const int i = blockIdx.x * 256 + threadIdx.x;
  Wh[i] = (_Float16)W[i];
}

// ---- GEMM1 + fused WTA: per row, rowKey = max over n of pack(residual, n) ----
// Block = 4 waves = 64 M rows x 32 N cols. B chunk (32 cols x 32 K halves, 2KB)
// staged in LDS via TDM (double buffered); each wave does 2 wmma per K-step.
__global__ __launch_bounds__(128)
void k_gemm_wta(const _Float16* __restrict__ Ah,       // [NB x INF] f16
                const _Float16* __restrict__ Wh,       // [OUTF x INF] f16
                const float*    __restrict__ encoded,  // [NB x OUTF]
                const int*      __restrict__ maxNZ,
                unsigned long long* __restrict__ rowKey) {
  __shared__ _Float16 Bs[2][32 * 32];    // 2 x 2KB double buffer
  const int lane  = threadIdx.x & 31;
  const int wave  = threadIdx.x >> 5;
  const int mTile = blockIdx.y * 64 + wave * 16;
  const int nBase = blockIdx.x * 32;

  // A fragment (16x32 f16): lane m = lane&15; 16 contiguous K-halves {aOff..+15}
  const int aRow = mTile + (lane & 15);
  const int aOff = (lane & 16) ? 16 : 0;
  const _Float16* aPtr = Ah + (size_t)aRow * INF + aOff;
  // B fragment: local col = lane&15 (+16 for 2nd tile); 16 contiguous K-halves at bOff
  const int bRow = lane & 15;
  const int bOff = (lane & 16) ? 16 : 0;

  v8f c0 = {}, c1 = {};

#if HAS_TDM
  if (wave == 0)
    tdm_load_B(Wh + (size_t)nBase * INF, lds_offset(&Bs[0][0]));
#endif

  for (int kk = 0; kk < KSTEPS; ++kk) {
    const int cur = kk & 1;
    const int k0  = kk * 32;
#if HAS_TDM
    if (wave == 0) {
      if (kk + 1 < KSTEPS) {
        tdm_load_B(Wh + (size_t)nBase * INF + (k0 + 32),
                   lds_offset(&Bs[cur ^ 1][0]));
        __builtin_amdgcn_s_wait_tensorcnt(1);   // current buffer's TDM done
      } else {
        __builtin_amdgcn_s_wait_tensorcnt(0);
      }
    }
    __syncthreads();
#else
    {
      const int r = threadIdx.x >> 2, seg = threadIdx.x & 3;
      *(v8h*)&Bs[cur][r * 32 + seg * 8] =
          *(const v8h*)(Wh + (size_t)(nBase + r) * INF + k0 + seg * 8);
    }
    __syncthreads();
#endif
    if (k0 + 32 < INF) __builtin_prefetch(aPtr + k0 + 32, 0, 0);
    v8h alo = *(const v8h*)(aPtr + k0);
    v8h ahi = *(const v8h*)(aPtr + k0 + 8);
    v16h a;
    #pragma unroll
    for (int h = 0; h < 8; ++h) { a[h] = alo[h]; a[h + 8] = ahi[h]; }
    v16h b0 = *(const v16h*)&Bs[cur][bRow * 32 + bOff];
    v16h b1 = *(const v16h*)&Bs[cur][(bRow + 16) * 32 + bOff];
    c0 = __builtin_amdgcn_wmma_f32_16x16x32_f16(false, a, false, b0, (short)0, c0, false, false);
    c1 = __builtin_amdgcn_wmma_f32_16x16x32_f16(false, a, false, b1, (short)0, c1, false, false);
    __syncthreads();   // all reads of 'cur' done before it is refilled
  }

  const float lambd = 3.0f * (float)(*maxNZ) * (float)INF;
  // C/D layout: VGPR r -> M = r + 8*(lane>=16), N = lane&15
  const int n0 = nBase + (lane & 15);
  const int n1 = n0 + 16;
  const int mB = mTile + ((lane & 16) ? 8 : 0);
  #pragma unroll
  for (int r = 0; r < 8; ++r) {
    const int m  = mB + r;
    const float v0 = c0[r] - lambd * encoded[(size_t)m * OUTF + n0];
    const float v1 = c1[r] - lambd * encoded[(size_t)m * OUTF + n1];
    unsigned long long ka = pack_key(v0, n0);
    unsigned long long kb = pack_key(v1, n1);
    unsigned long long key = ka > kb ? ka : kb;
    #pragma unroll
    for (int off = 1; off < 16; off <<= 1) {         // reduce within 16-lane half
      unsigned long long o = __shfl_xor(key, off, 32);
      if (o > key) key = o;
    }
    if ((lane & 15) == 0) atomicMax(&rowKey[m], key);
  }
}

// ---- decode WTA winner + sparse GEMM2 + top-6 kwta + rebuild A ----
__global__ __launch_bounds__(256)
void k_gather_topk(float* __restrict__ encoded,            // [NB x OUTF] (updated)
                   unsigned long long* __restrict__ rowKey,
                   const float* __restrict__ W,            // [OUTF x INF] f32
                   const float* __restrict__ x,            // [NB x INF]
                   float* __restrict__ xr,                 // [NB x INF]
                   _Float16* __restrict__ Ah) {            // [NB x INF]
  const int row  = blockIdx.x * 8 + (threadIdx.x >> 5);
  const int lane = threadIdx.x & 31;

  const unsigned long long key = rowKey[row];
  const unsigned nNew = 0xFFFFFFFFu - (unsigned)(key & 0xFFFFFFFFull);
  if (lane == 0) rowKey[row] = 0ull;   // reset for next step

  float acc[16];
  #pragma unroll
  for (int c = 0; c < 16; ++c) acc[c] = 0.0f;

  float eStore = 0.0f;
  const float* erow = encoded + (size_t)row * OUTF;
  for (int o0 = 0; o0 < OUTF; o0 += 32) {
    float e = erow[o0 + lane];
    if ((unsigned)(o0 + lane) == nNew) { e += 1.0f; eStore = e; }  // apply WTA in-reg
    unsigned act = __builtin_amdgcn_ballot_w32(e != 0.0f);
    while (act) {
      const int src = __builtin_ctz(act);
      act &= act - 1;
      const float ev = __shfl(e, src, 32);
      const float* wrow = W + (size_t)(o0 + src) * INF;
      #pragma unroll
      for (int c = 0; c < 16; ++c) acc[c] += ev * wrow[c * 32 + lane];
    }
  }
  if ((int)(nNew & 31u) == lane)       // persist WTA update for next GEMM
    encoded[(size_t)row * OUTF + nNew] = eStore;

  // iterative top-6 over 512 values (lane holds cols c*32+lane)
  unsigned winMask = 0;
  #pragma unroll
  for (int t = 0; t < KIN; ++t) {
    float best = -__builtin_inff();
    int   bi   = 0;
    #pragma unroll
    for (int c = 0; c < 16; ++c)
      if (acc[c] > best) { best = acc[c]; bi = c * 32 + lane; }
    wave_argmax(best, bi);
    if ((bi & 31) == lane) { acc[bi >> 5] = -__builtin_inff(); winMask |= 1u << (bi >> 5); }
  }

  const size_t base = (size_t)row * INF;
  #pragma unroll
  for (int c = 0; c < 16; ++c) {
    const int col = c * 32 + lane;
    const float xv = ((winMask >> c) & 1u) ? 1.0f : 0.0f;
    xr[base + col] = xv;
    Ah[base + col] = (_Float16)(2.0f * x[base + col] - xv);
  }
}

// ---------------- host side -------------------------------------------------
extern "C" void kernel_launch(void* const* d_in, const int* in_sizes, int n_in,
                              void* d_out, int out_size, void* d_ws, size_t ws_size,
                              hipStream_t stream) {
  const float* x = (const float*)d_in[0];   // [4096 x 512] binary f32
  const float* W = (const float*)d_in[1];   // [1024 x 512] f32

  float* out     = (float*)d_out;
  float* encoded = out;                      // [4096 x 1024]
  float* xr      = out + (size_t)NB * OUTF;  // [4096 x 512]

  char* ws = (char*)d_ws;
  _Float16* Ah     = (_Float16*)ws;                                   // 4 MB
  _Float16* Wh     = (_Float16*)(ws + (size_t)4 * 1024 * 1024);       // 1 MB
  unsigned long long* rowKey =
      (unsigned long long*)(ws + (size_t)5 * 1024 * 1024);            // 32 KB
  int* maxNZ = (int*)(ws + (size_t)5 * 1024 * 1024 + 64 * 1024);

  k_init_misc<<<1, 1, 0, stream>>>(maxNZ);
  k_init_keys<<<NB / 256, 256, 0, stream>>>(rowKey);
  k_count_rows<<<NB / 8, 256, 0, stream>>>(x, maxNZ);
  k_init_enc<<<(NB * OUTF) / 256, 256, 0, stream>>>(encoded);
  k_init_x<<<(NB * INF) / 256, 256, 0, stream>>>(x, xr, Ah);
  k_conv_w<<<(OUTF * INF) / 256, 256, 0, stream>>>(W, Wh);

  for (int s = 0; s < KACT; ++s) {
    k_gemm_wta<<<dim3(OUTF / 32, NB / 64), 128, 0, stream>>>(
        Ah, Wh, encoded, maxNZ, rowKey);
    k_gather_topk<<<NB / 8, 256, 0, stream>>>(encoded, rowKey, W, x, xr, Ah);
  }
}
